// MLayer_29832842838337
// MI455X (gfx1250) — compile-verified, matrix operand order checked by reading, not run on hardware
//
#include <hip/hip_runtime.h>

#define KDIM 768
#define MATN 256      // 16*16 output elems per matrix
#define KC_COUNT 24   // 768 / 32
#define TST 20        // transposed-copy row stride (floats): 20*l mod 64 distinct for l<16,
                      // rows are 80B -> 16B-aligned b128 writebacks
#define MSTT (16 * TST)

typedef __bf16 v16bf __attribute__((ext_vector_type(16)));
typedef __bf16 v2bf  __attribute__((ext_vector_type(2)));
typedef unsigned short v16us __attribute__((ext_vector_type(16)));
typedef unsigned int v8u __attribute__((ext_vector_type(8)));
typedef float v8f __attribute__((ext_vector_type(8)));
typedef float v2f __attribute__((ext_vector_type(2)));

__device__ __forceinline__ unsigned short f32_to_bf16_rne(float f) {
  unsigned int u = __builtin_bit_cast(unsigned int, f);
  u += 0x7FFFu + ((u >> 16) & 1u);
  return (unsigned short)(u >> 16);
}

// Pre-pack W [768][256] f32 -> bf16 in the exact WMMA B-operand lane/half layout:
// dst[((kc*16+nt)*32 + lane)*16 + h] = bf16( W[k][n] ),
//   k = kc*32 + (h&7) + 16*(h>>3) + 8*(lane>>4),  n = nt*16 + (lane&15)
__global__ void pack_W_bf16(const float* __restrict__ W, unsigned short* __restrict__ Bp) {
  int t = blockIdx.x * blockDim.x + threadIdx.x;
  if (t >= KDIM * MATN) return;
  int h    = t & 15;
  int lane = (t >> 4) & 31;
  int nt   = (t >> 9) & 15;
  int kc   = t >> 13;
  int k = kc * 32 + (h & 7) + ((h >> 3) << 4) + ((lane >> 4) << 3);
  int n = nt * 16 + (lane & 15);
  Bp[t] = f32_to_bf16_rne(W[k * MATN + n]);
}

// One wave (32 threads) per 16 rows of x == 16 output matrices.
__global__ __launch_bounds__(32)
void matexp_kernel(const float* __restrict__ x,
                   const unsigned short* __restrict__ Bp,
                   const float* __restrict__ bias,
                   float* __restrict__ out) {
  // Transposed matrices only: T_m[j][i] = mat_m[i][j]
  __shared__ float T[16 * MSTT];   // 16 * 320 * 4 = 20 KB

  const int lane = threadIdx.x & 31;
  const int lj   = lane & 15;   // col / row-within-halfwave index
  const int lh   = lane >> 4;   // half-wave select
  const int row0 = blockIdx.x * 16;

  v8f acc[16];
#pragma unroll
  for (int nt = 0; nt < 16; ++nt) acc[nt] = v8f{};

  const float* xrow = x + (size_t)(row0 + lj) * KDIM;

  // ---------------- GEMM: [16 x 768] @ [768 x 256], bf16 WMMA ----------------
  for (int kc = 0; kc < KC_COUNT; ++kc) {
    // A fragment 16x32 bf16: lane<16 holds row lj, K = kc*32 + {0..7, 16..23};
    // lane>=16 holds row lj, K = kc*32 + {8..15, 24..31}.
    const int kb = kc * 32 + lh * 8;
    float4 a0 = *(const float4*)(xrow + kb);
    float4 a1 = *(const float4*)(xrow + kb + 4);
    float4 a2 = *(const float4*)(xrow + kb + 16);
    float4 a3 = *(const float4*)(xrow + kb + 20);
    float af[16] = {a0.x, a0.y, a0.z, a0.w, a1.x, a1.y, a1.z, a1.w,
                    a2.x, a2.y, a2.z, a2.w, a3.x, a3.y, a3.z, a3.w};
    union { v16us u; v16bf b; unsigned int u32[8]; } A;
#if __has_builtin(__builtin_amdgcn_cvt_pk_bf16_f32)
#pragma unroll
    for (int i = 0; i < 8; ++i) {
      v2bf p = __builtin_amdgcn_cvt_pk_bf16_f32(af[2 * i], af[2 * i + 1]);
      A.u32[i] = __builtin_bit_cast(unsigned int, p);
    }
#else
#pragma unroll
    for (int i = 0; i < 16; ++i) A.u[i] = f32_to_bf16_rne(af[i]);
#endif

#pragma unroll
    for (int nt = 0; nt < 16; ++nt) {
      union { v8u u; v16bf b; } B;
      B.u = *(const v8u*)(Bp + (((size_t)(kc * 16 + nt) * 32 + lane) << 4));
      acc[nt] = __builtin_amdgcn_wmma_f32_16x16x32_bf16(
          false, A.b, false, B.b, (short)0, acc[nt], false, false);
    }
  }

  // ------------- epilogue: + bias, * 2^-4, + I ; spill transposed to LDS -------------
  // C layout: VGPR v, lane l -> matrix m = v + 8*lh, element [i=nt][j=lj].
  // T_m[j][i]: addr = m*MSTT + lj*TST + nt -> explicit float4 stores (16B aligned).
  float bias_r[16];
#pragma unroll
  for (int nt = 0; nt < 16; ++nt) bias_r[nt] = bias[nt * 16 + lj];

#pragma unroll
  for (int v = 0; v < 8; ++v) {
    float* Tm = &T[(v + lh * 8) * MSTT + lj * TST];
#pragma unroll
    for (int nt4 = 0; nt4 < 4; ++nt4) {
      float4 s;
      {
        int nt = nt4 * 4 + 0;
        s.x = (acc[nt][v] + bias_r[nt]) * 0.0625f + ((nt == lj) ? 1.0f : 0.0f);
      }
      {
        int nt = nt4 * 4 + 1;
        s.y = (acc[nt][v] + bias_r[nt]) * 0.0625f + ((nt == lj) ? 1.0f : 0.0f);
      }
      {
        int nt = nt4 * 4 + 2;
        s.z = (acc[nt][v] + bias_r[nt]) * 0.0625f + ((nt == lj) ? 1.0f : 0.0f);
      }
      {
        int nt = nt4 * 4 + 3;
        s.w = (acc[nt][v] + bias_r[nt]) * 0.0625f + ((nt == lj) ? 1.0f : 0.0f);
      }
      *(float4*)(Tm + nt4 * 4) = s;   // ds_store_b128
    }
  }

  // ---------------- 4 squaring rounds: mat = mat @ mat, f32 WMMA ----------------
  for (int r = 0; r < 4; ++r) {
    for (int m = 0; m < 16; ++m) {
      float* Tm = &T[m * MSTT];

      // A-fragment gathers: mat[lj][k] = T[k][lj]; common base + small immediate
      // deltas -> pairs into ds_load_2addr_b32.
      float a_el[8];
#pragma unroll
      for (int q = 0; q < 4; ++q) {
        a_el[2 * q + 0] = Tm[(q * 4 + 2 * lh + 0) * TST + lj];
        a_el[2 * q + 1] = Tm[(q * 4 + 2 * lh + 1) * TST + lj];
      }

      v8f c = v8f{};
#pragma unroll
      for (int q = 0; q < 4; ++q) {
        // A 16x4: lane<16: row=lj, K = q*4 + {0,1}; lane>=16: K = q*4 + {2,3}
        // B 4x16: lane<16: col=lj, K = q*4 + {0,1}; lane>=16: K = q*4 + {2,3}
        v2f a;
        a.x = a_el[2 * q + 0];
        a.y = a_el[2 * q + 1];
        v2f b = *(const v2f*)(Tm + lj * TST + q * 4 + 2 * lh);  // ds_load_b64
        c = __builtin_amdgcn_wmma_f32_16x16x4_f32(
            false, a, false, b, (short)0, c, false, false);
      }

      if (r < 3) {
        // Writeback transposed: T[lj][v+8*lh] = c[v] -> two explicit b128 stores
        // (byte addr lj*80 + lh*32 -> 16B aligned).
        float4 s0 = {c[0], c[1], c[2], c[3]};
        float4 s1 = {c[4], c[5], c[6], c[7]};
        float* wb = Tm + lj * TST + lh * 8;
        *(float4*)(wb + 0) = s0;
        *(float4*)(wb + 4) = s1;
      } else {
        float* op = out + (size_t)(row0 + m) * MATN;
#pragma unroll
        for (int v = 0; v < 8; ++v) op[(v + lh * 8) * 16 + lj] = c[v];
      }
    }
  }
}

extern "C" void kernel_launch(void* const* d_in, const int* in_sizes, int n_in,
                              void* d_out, int out_size, void* d_ws, size_t ws_size,
                              hipStream_t stream) {
  const float* x    = (const float*)d_in[0];
  const float* W    = (const float*)d_in[1];
  const float* bias = (const float*)d_in[2];
  float* out = (float*)d_out;
  unsigned short* Bp = (unsigned short*)d_ws;  // needs 768*256*2 = 384 KiB

  int packElems = KDIM * MATN;
  pack_W_bf16<<<(packElems + 255) / 256, 256, 0, stream>>>(W, Bp);

  int nRows   = in_sizes[0] / KDIM;  // 65536
  int nBlocks = nRows / 16;          // one wave per 16 matrices
  matexp_kernel<<<nBlocks, 32, 0, stream>>>(x, Bp, bias, out);
}
